// AutoCorrelationAttention_84997402788068
// MI455X (gfx1250) — matrix operand order repaired and checked by reading, not compiled, not run on hardware
//
#include <hip/hip_runtime.h>

typedef __attribute__((ext_vector_type(16))) _Float16 v16h;
typedef __attribute__((ext_vector_type(8)))  _Float16 h8;
typedef __attribute__((ext_vector_type(8)))  float    v8f;

#define LSEQ     4096
#define LMASK    4095
#define L32      2048
#define L32MASK  2047
#define DCH      64
#define KTOP     24
#define NTHREADS 128

// One block (4 wave32) per (b,d) channel pair. Fused:
//   WMMA circular correlation -> top-24 -> softmax -> weighted circular V gather.
// Correlation tile: D[m,n] = R[16m + 256n + r] for r = 0..15, with
//   A_j[m,k] = Qr[(32j + 16m + k) & 4095]   (Qr = Q row rotated by r)
//   B_j[k,n] = Kh[(32j + k - 256n) & 4095]
// Both operands' per-lane LDS runs are 16B-aligned -> pure ds_load_b128 feeds.
__global__ __launch_bounds__(NTHREADS)
void autocorr_attn_wmma_kernel(const float* __restrict__ Q,
                               const float* __restrict__ K,
                               const float* __restrict__ V,
                               float* __restrict__ Out)
{
    __shared__ alignas(16) unsigned int Qh32[L32];   // f16 Q row (u32-packed)
    __shared__ alignas(16) unsigned int Kh32[L32];   // f16 K row
    __shared__ alignas(16) unsigned int Qr32[L32];   // rotated Q row
    __shared__ float Rsh[LSEQ];                      // correlation scores
    __shared__ float Vrow[LSEQ];                     // f32 V row
    __shared__ float s_val[NTHREADS];
    __shared__ int   s_idx[NTHREADS];
    __shared__ float topv[KTOP];
    __shared__ int   topi[KTOP];

    const int tid = threadIdx.x;
    const int b   = blockIdx.x >> 6;    // / DCH
    const int d   = blockIdx.x & 63;    // % DCH
    const size_t rowbase = ((size_t)b * LSEQ) * DCH + (size_t)d;

    // ---- Phase 1: stage Q,K rows as packed f16 pairs; zero scores ----
    for (int i = tid; i < L32; i += NTHREADS) {
        float q0 = Q[rowbase + (size_t)(2 * i)     * DCH];
        float q1 = Q[rowbase + (size_t)(2 * i + 1) * DCH];
        float k0 = K[rowbase + (size_t)(2 * i)     * DCH];
        float k1 = K[rowbase + (size_t)(2 * i + 1) * DCH];
        union { _Float16 h[2]; unsigned int u; } pq, pk;
        pq.h[0] = (_Float16)q0; pq.h[1] = (_Float16)q1;
        pk.h[0] = (_Float16)k0; pk.h[1] = (_Float16)k1;
        Qh32[i] = pq.u;
        Kh32[i] = pk.u;
    }
    for (int i = tid; i < LSEQ; i += NTHREADS) Rsh[i] = 0.0f;
    __syncthreads();

    // ---- Phase 2: WMMA correlation ----
    const int  lane  = tid & 31;
    const int  wv    = tid >> 5;          // wave 0..3
    const int  mrow  = lane & 15;
    const bool hiH   = (lane >= 16);
    // A as fed here (contiguous k): lanes 0-15: M=lane, K=e; lanes 16-31: M=lane-16, K=e+16
    const int  abase = (mrow << 4) + (hiH ? 16 : 0);
    // B as fed here (contiguous k): lanes 0-15: n=lane, K=e; lanes 16-31: n=lane-16, K=e+16
    const int  bconst = (hiH ? 16 : 0) - (mrow << 8);
    const _Float16* Qr = (const _Float16*)Qr32;
    const _Float16* Kh = (const _Float16*)Kh32;
    const int  j0 = wv << 5;              // each wave owns 32 of the 128 j-steps

    for (int r = 0; r < 16; ++r) {
        // Rotate Q row by r (u32 view: dword copy for even r, funnel shift for odd r)
        const int s2 = r >> 1;
        if ((r & 1) == 0) {
            for (int i = tid; i < L32; i += NTHREADS)
                Qr32[i] = Qh32[(i + s2) & L32MASK];
        } else {
            for (int i = tid; i < L32; i += NTHREADS) {
                unsigned lo = Qh32[(i + s2)     & L32MASK];
                unsigned hh = Qh32[(i + s2 + 1) & L32MASK];
                Qr32[i] = (lo >> 16) | (hh << 16);
            }
        }
        __syncthreads();

        v8f acc = {};
        for (int jj = 0; jj < 32; ++jj) {
            const int j     = j0 + jj;
            const int afull = (j << 5) + abase;          // == 0 mod 8
            const int p1    = afull & LMASK;             // 8-half run, 16B aligned
            const int p2    = (afull + 8) & LMASK;       // 8-half run, 16B aligned
            const int p3    = ((j << 5) + bconst) & LMASK; // == 0 mod 16
            h8 alo = *(const h8*)(Qr + p1);
            h8 ahi = *(const h8*)(Qr + p2);
            h8 blo = *(const h8*)(Kh + p3);
            h8 bhi = *(const h8*)(Kh + p3 + 8);
            v16h av = __builtin_shufflevector(alo, ahi, 0,1,2,3,4,5,6,7,8,9,10,11,12,13,14,15);
            v16h bv = __builtin_shufflevector(blo, bhi, 0,1,2,3,4,5,6,7,8,9,10,11,12,13,14,15);
            acc = __builtin_amdgcn_wmma_f32_16x16x32_f16(
                      false, av, false, bv, (short)0, acc, false, false);
        }
        // C/D layout: VGPR ri -> M = ri (+8 for lanes 16-31), N = lane & 15
#pragma unroll
        for (int ri = 0; ri < 8; ++ri) {
            const int m   = ri + (hiH ? 8 : 0);
            const int tau = (m << 4) + (mrow << 8) + r;
            atomicAdd(&Rsh[tau], acc[ri]);    // combine the 4 waves' j-partials
        }
        __syncthreads();
    }

    // ---- Phase 3: top-24 by iterative block-wide argmax ----
    for (int kk = 0; kk < KTOP; ++kk) {
        float best = -3.0e38f; int bi = 0;
        for (int i = tid; i < LSEQ; i += NTHREADS) {
            float v = Rsh[i];
            if (v > best) { best = v; bi = i; }
        }
        s_val[tid] = best; s_idx[tid] = bi;
        __syncthreads();
        for (int s = NTHREADS >> 1; s > 0; s >>= 1) {
            if (tid < s && s_val[tid + s] > s_val[tid]) {
                s_val[tid] = s_val[tid + s];
                s_idx[tid] = s_idx[tid + s];
            }
            __syncthreads();
        }
        if (tid == 0) {
            topv[kk] = s_val[0];
            topi[kk] = s_idx[0];
            Rsh[s_idx[0]] = -3.0e38f;   // mask out for next round
        }
        __syncthreads();
    }

    // ---- Phase 4: softmax over 24 scores (thread 0) ----
    if (tid == 0) {
        float mx = topv[0];
        for (int kk = 1; kk < KTOP; ++kk) mx = fmaxf(mx, topv[kk]);
        float ssum = 0.f;
        for (int kk = 0; kk < KTOP; ++kk) {
            float e = __expf(topv[kk] - mx);
            topv[kk] = e; ssum += e;
        }
        float inv = 1.f / ssum;
        for (int kk = 0; kk < KTOP; ++kk) topv[kk] *= inv;
    }
    // ---- Phase 5: stage V row (overlaps thread-0 softmax) ----
    for (int i = tid; i < LSEQ; i += NTHREADS) {
        Vrow[i] = V[rowbase + (size_t)i * DCH];
    }
    __syncthreads();

    // ---- Phase 6: out[l] = sum_k w_k * V[(l + i_k) % L] ----
    for (int l = tid; l < LSEQ; l += NTHREADS) {
        float acco = 0.f;
#pragma unroll 4
        for (int kk = 0; kk < KTOP; ++kk) {
            acco = fmaf(topv[kk], Vrow[(l + topi[kk]) & LMASK], acco);
        }
        Out[rowbase + (size_t)l * DCH] = acco;
    }
}

extern "C" void kernel_launch(void* const* d_in, const int* in_sizes, int n_in,
                              void* d_out, int out_size, void* d_ws, size_t ws_size,
                              hipStream_t stream) {
    const float* Q = (const float*)d_in[0];
    const float* K = (const float*)d_in[1];
    const float* V = (const float*)d_in[2];
    float* Out = (float*)d_out;

    const int Btot = in_sizes[0] / (LSEQ * DCH);   // = 16 for the reference shapes
    dim3 grid((unsigned)(Btot * DCH));
    dim3 block(NTHREADS);
    hipLaunchKernelGGL(autocorr_attn_wmma_kernel, grid, block, 0, stream, Q, K, V, Out);
}
